// GumbelQuantize_68796786147424
// MI455X (gfx1250) — matrix operand order repaired and checked by reading, not compiled, not run on hardware
//
#include <hip/hip_runtime.h>
#include <hip/hip_bf16.h>

// ---------------------------------------------------------------------------
// Gumbel-softmax VQ, flash-attention style fusion for gfx1250 (MI455X).
//   B*L = M = 8192 rows, D = 1024, C = 8192 codes.
//   Pass A: pack bf16 copies of Q and E (+ transposed E) into workspace.
//   Pass B: fused  S = Q E^T  ->  +gumbel -> online softmax -> O = P E
//           using v_wmma_f32_16x16x32_bf16; MT=32 rows so each B fragment
//           feeds two WMMAs; Q tile staged with global_load_async_to_lds_b128
//           (inline asm) + s_wait_asynccnt.
// ---------------------------------------------------------------------------

typedef __attribute__((ext_vector_type(16))) __bf16 v16bf;
typedef __attribute__((ext_vector_type(8)))  float  v8f;

#define M_TOT 8192
#define D_DIM 1024
#define C_DIM 8192
#define CT    128          // C tile per iteration (16 cols per wave in GEMM1)
#define MT    32           // M rows per workgroup (2 x 16-row WMMA tiles)
#define NWARP 8

// Async Q-tile staging via CDNA5 async-to-LDS path (inline asm, ASYNCcnt).
#define USE_ASYNC_LDS_ASM 1

union Frag16 { uint4 u[2]; v16bf v; };   // 16 bf16 = 32B = one WMMA A/B fragment

__device__ __forceinline__ unsigned short f32_to_bf16_rte(float f) {
    unsigned int u = __float_as_uint(f);
    unsigned int r = u + 0x7FFFu + ((u >> 16) & 1u);   // round-to-nearest-even
    return (unsigned short)(r >> 16);
}

// ---------------------------------------------------------------------------
// Pass A: f32 -> bf16 packing.  Q16[M][D], E16[C][D], ET16[D][C] (transposed).
// ---------------------------------------------------------------------------
__global__ void GumbelQuantize_pack_kernel(const float* __restrict__ mlc,
                                           const float* __restrict__ emb,
                                           unsigned short* __restrict__ Q16,
                                           unsigned short* __restrict__ E16,
                                           unsigned short* __restrict__ ET16) {
    const size_t n = (size_t)M_TOT * D_DIM;   // == C_DIM * D_DIM
    for (size_t i = (size_t)blockIdx.x * blockDim.x + threadIdx.x; i < n;
         i += (size_t)gridDim.x * blockDim.x) {
        Q16[i] = f32_to_bf16_rte(mlc[i]);
        const unsigned short eb = f32_to_bf16_rte(emb[i]);
        E16[i] = eb;
        const size_t c = i >> 10;       // / D_DIM
        const size_t d = i & 1023;      // % D_DIM
        ET16[d * (size_t)C_DIM + c] = eb;
    }
}

// ---------------------------------------------------------------------------
// Pass B: fused VQ.  One workgroup = 32 rows, 8 waves.
//   GEMM1: wave w -> S columns [c0+16w, +16) for both row halves (64 WMMAs)
//   GEMM2: wave w -> output D-slice [128w, +128) for both row halves (64 WMMAs)
// ---------------------------------------------------------------------------
__global__ __launch_bounds__(256, 1)
void GumbelQuantize_fused_kernel(const unsigned short* __restrict__ Q16,
                                 const unsigned short* __restrict__ E16,
                                 const unsigned short* __restrict__ ET16,
                                 const float* __restrict__ gumbel_u,
                                 const float* __restrict__ tau_p,
                                 float* __restrict__ quant_out,
                                 float* __restrict__ codes_out) {
    __shared__ unsigned short Qs[MT * D_DIM];   // 64 KB: Q tile, bf16 row-major
    __shared__ float          Ss[MT * CT];      // 16 KB: noisy scaled logits
    __shared__ unsigned short Ps[MT * CT];      //  8 KB: exp(s - m), bf16
    __shared__ float row_m[MT], row_l[MT], row_alpha[MT], row_best[MT];
    __shared__ int   row_bidx[MT];

    const int tid  = threadIdx.x;
    const int wave = tid >> 5;
    const int lane = tid & 31;
    const int lmod = lane & 15;     // row (A) / column (B,C,D) within fragment
    const int half = lane >> 4;     // K-half selector per ISA 7.12.2 layouts
    const int m0   = blockIdx.x * MT;

    // ---- Stage Q tile (contiguous 64KB) via async copy to LDS. -------------
    {
        const uint4* src = reinterpret_cast<const uint4*>(Q16 + (size_t)m0 * D_DIM);
        uint4*       dst = reinterpret_cast<uint4*>(Qs);
#ifdef USE_ASYNC_LDS_ASM
        for (int i = tid; i < (MT * D_DIM) / 8; i += 256) {
            const uint4* g = src + i;
            // Generic LDS pointer = {shared aperture, 32-bit LDS offset};
            // the async op's VDST takes the 32-bit LDS byte address.
            unsigned lds_addr = (unsigned)(uintptr_t)(dst + i);
            asm volatile("global_load_async_to_lds_b128 %0, %1, off"
                         :
                         : "v"(lds_addr), "v"(g)
                         : "memory");
        }
        asm volatile("s_wait_asynccnt 0x0" ::: "memory");
#else
        for (int i = tid; i < (MT * D_DIM) / 8; i += 256) dst[i] = src[i];
#endif
    }
    if (tid < MT) {
        row_m[tid]     = -3.0e38f;
        row_l[tid]     = 0.0f;
        row_alpha[tid] = 0.0f;
        row_best[tid]  = -3.0e38f;
        row_bidx[tid]  = 0;
    }
    const float inv_tau = 1.0f / tau_p[0];

    v8f acc0[8], acc1[8];            // O slices for row halves: 128 VGPRs total
    #pragma unroll
    for (int t = 0; t < 8; ++t)
        #pragma unroll
        for (int r = 0; r < 8; ++r) { acc0[t][r] = 0.0f; acc1[t][r] = 0.0f; }

    __syncthreads();

    for (int ct = 0; ct < C_DIM / CT; ++ct) {
        const int c0 = ct * CT;
        const int cw = c0 + wave * 16;

        // Prefetch next C-tile of the codebook (global_prefetch_b8).
        if (ct + 1 < C_DIM / CT && lane < 16) {
            __builtin_prefetch(E16 + ((size_t)(c0 + CT) + wave * 16 + lane) * D_DIM, 0, 0);
        }

        // -------- GEMM1: S = Q @ E_tile^T.  B fragment shared by 2 WMMAs. ----
        v8f s0, s1;
        #pragma unroll
        for (int r = 0; r < 8; ++r) { s0[r] = 0.0f; s1[r] = 0.0f; }

        const unsigned short* qrow0 = &Qs[lmod * D_DIM];
        const unsigned short* qrow1 = &Qs[(16 + lmod) * D_DIM];
        const unsigned short* erow  = E16 + (size_t)(cw + lmod) * D_DIM;

        #pragma unroll 2
        for (int kb = 0; kb < D_DIM; kb += 32) {
            Frag16 a0, a1, b;
            a0.u[0] = *reinterpret_cast<const uint4*>(qrow0 + kb + half * 8);
            a0.u[1] = *reinterpret_cast<const uint4*>(qrow0 + kb + 16 + half * 8);
            a1.u[0] = *reinterpret_cast<const uint4*>(qrow1 + kb + half * 8);
            a1.u[1] = *reinterpret_cast<const uint4*>(qrow1 + kb + 16 + half * 8);
            const uint4* ep = reinterpret_cast<const uint4*>(erow + kb + half * 8);
            b.u[0] = ep[0];
            b.u[1] = ep[2];
            s0 = __builtin_amdgcn_wmma_f32_16x16x32_bf16(false, a0.v, false, b.v,
                                                         (short)0, s0, false, false);
            s1 = __builtin_amdgcn_wmma_f32_16x16x32_bf16(false, a1.v, false, b.v,
                                                         (short)0, s1, false, false);
        }

        // -------- gumbel noise + 1/tau, spill S' into LDS (C/D frag layout) --
        #pragma unroll
        for (int r = 0; r < 8; ++r) {
            const int mrow = r + 8 * half;
            const int gc   = cw + lmod;
            float u0 = gumbel_u[(size_t)(m0 + mrow) * C_DIM + gc];
            float u1 = gumbel_u[(size_t)(m0 + 16 + mrow) * C_DIM + gc];
            u0 = fmaxf(u0, 1e-10f);
            u1 = fmaxf(u1, 1e-10f);
            const float g0 = -__logf(-__logf(u0));
            const float g1 = -__logf(-__logf(u1));
            Ss[mrow * CT + wave * 16 + lmod]        = (s0[r] + g0) * inv_tau;
            Ss[(16 + mrow) * CT + wave * 16 + lmod] = (s1[r] + g1) * inv_tau;
        }
        __syncthreads();

        // -------- online softmax state per row (one wave, 32 rows) -----------
        if (tid < MT) {
            const int row = tid;
            float tmax = -3.0e38f; int targ = 0;
            for (int j = 0; j < CT; ++j) {
                const float v = Ss[row * CT + j];
                if (v > tmax) { tmax = v; targ = j; }
            }
            if (tmax > row_best[row]) { row_best[row] = tmax; row_bidx[row] = c0 + targ; }
            const float mold  = row_m[row];
            const float mnew  = fmaxf(mold, tmax);
            const float alpha = __expf(mold - mnew);
            float sum = 0.0f;
            for (int j = 0; j < CT; ++j) {
                const float p = __expf(Ss[row * CT + j] - mnew);
                sum += p;
                Ps[row * CT + j] = f32_to_bf16_rte(p);
            }
            row_l[row]     = row_l[row] * alpha + sum;
            row_m[row]     = mnew;
            row_alpha[row] = alpha;
        }
        __syncthreads();

        // -------- rescale O, then O += P @ E_tile (B shared by 2 WMMAs) ------
        float al0[8], al1[8];
        #pragma unroll
        for (int r = 0; r < 8; ++r) {
            al0[r] = row_alpha[r + 8 * half];
            al1[r] = row_alpha[16 + r + 8 * half];
        }
        #pragma unroll
        for (int t = 0; t < 8; ++t)
            #pragma unroll
            for (int r = 0; r < 8; ++r) { acc0[t][r] *= al0[r]; acc1[t][r] *= al1[r]; }

        const int dw = wave * 128;
        #pragma unroll
        for (int kb = 0; kb < CT; kb += 32) {
            Frag16 a0, a1;
            a0.u[0] = *reinterpret_cast<const uint4*>(&Ps[lmod * CT + kb + half * 8]);
            a0.u[1] = *reinterpret_cast<const uint4*>(&Ps[lmod * CT + kb + 16 + half * 8]);
            a1.u[0] = *reinterpret_cast<const uint4*>(&Ps[(16 + lmod) * CT + kb + half * 8]);
            a1.u[1] = *reinterpret_cast<const uint4*>(&Ps[(16 + lmod) * CT + kb + 16 + half * 8]);
            #pragma unroll
            for (int sub = 0; sub < 8; ++sub) {
                Frag16 b;   // B[k][n] = E[c0+kb+k][dw+16*sub+n] == ET16 row d
                const int d = dw + sub * 16 + lmod;
                const uint4* tp = reinterpret_cast<const uint4*>(
                    ET16 + (size_t)d * C_DIM + c0 + kb + half * 8);
                b.u[0] = tp[0];
                b.u[1] = tp[2];
                acc0[sub] = __builtin_amdgcn_wmma_f32_16x16x32_bf16(false, a0.v, false, b.v,
                                                                    (short)0, acc0[sub],
                                                                    false, false);
                acc1[sub] = __builtin_amdgcn_wmma_f32_16x16x32_bf16(false, a1.v, false, b.v,
                                                                    (short)0, acc1[sub],
                                                                    false, false);
            }
        }
        __syncthreads();   // Ss/Ps reused next iteration
    }

    // ---- finalize: O / l, write quantized + codes ---------------------------
    float linv0[8], linv1[8];
    #pragma unroll
    for (int r = 0; r < 8; ++r) {
        linv0[r] = 1.0f / row_l[r + 8 * half];
        linv1[r] = 1.0f / row_l[16 + r + 8 * half];
    }
    const int dw = wave * 128;
    #pragma unroll
    for (int sub = 0; sub < 8; ++sub) {
        const int d = dw + sub * 16 + lmod;
        #pragma unroll
        for (int r = 0; r < 8; ++r) {
            const int mrow = r + 8 * half;
            quant_out[(size_t)(m0 + mrow) * D_DIM + d]      = acc0[sub][r] * linv0[r];
            quant_out[(size_t)(m0 + 16 + mrow) * D_DIM + d] = acc1[sub][r] * linv1[r];
        }
    }
    if (tid < MT) codes_out[m0 + tid] = (float)row_bidx[tid];
}

// ---------------------------------------------------------------------------
extern "C" void kernel_launch(void* const* d_in, const int* in_sizes, int n_in,
                              void* d_out, int out_size, void* d_ws, size_t ws_size,
                              hipStream_t stream) {
    const float* mlc = (const float*)d_in[0];   // [8, 1024, 1024]
    const float* emb = (const float*)d_in[1];   // [8192, 1024]
    const float* gum = (const float*)d_in[2];   // [8, 1024, 8192]
    const float* tau = (const float*)d_in[3];   // scalar

    unsigned short* Q16  = (unsigned short*)d_ws;                    // 16 MB
    unsigned short* E16  = Q16 + (size_t)M_TOT * D_DIM;              // 16 MB
    unsigned short* ET16 = E16 + (size_t)C_DIM * D_DIM;              // 16 MB

    float* qout  = (float*)d_out;                                    // [M, D]
    float* codes = qout + (size_t)M_TOT * D_DIM;                     // [M] as f32

    GumbelQuantize_pack_kernel<<<2048, 256, 0, stream>>>(mlc, emb, Q16, E16, ET16);
    GumbelQuantize_fused_kernel<<<M_TOT / MT, 256, 0, stream>>>(Q16, E16, ET16,
                                                                gum, tau, qout, codes);
}
